// CrossAttentionBlock_47339129536604
// MI455X (gfx1250) — matrix operand order repaired and checked by reading, not compiled, not run on hardware
//
#include <hip/hip_runtime.h>
#include <hip/hip_bf16.h>

// CDNA5 / gfx1250 flash-attention for the CrossAttentionBlock reference.
// B=4, C=256, H=W=64 (HW=4096), d=32.  All matmuls on v_wmma_f32_16x16x32_bf16.
// K/V key-blocks double-buffered in LDS via global_load_async_to_lds_b128
// (ASYNCcnt); padded LDS strides keep all ds_read_b128 operand fetches
// bank-conflict free.

#define BB 4
#define CC 256
#define HW 4096
#define DD 32

// Padded LDS row strides (bf16 elements): lane offsets (20/36 dwords) are all
// distinct mod 64 banks -> conflict-free b128 reads.
#define KSTR 40   // sK rows of 32 data bf16
#define VSTR 72   // sV rows of 64 data bf16
#define PSTR 72   // sP rows of 64 data bf16

#if defined(__HIP_DEVICE_COMPILE__) && \
    __has_builtin(__builtin_amdgcn_global_load_async_to_lds_b128)
#define HAVE_ASYNC 1
#else
#define HAVE_ASYNC 0
#endif

#if HAVE_ASYNC
#if __has_builtin(__builtin_amdgcn_s_wait_asynccnt)
#define WAIT_ASYNC(n) __builtin_amdgcn_s_wait_asynccnt(n)
#else
#define WAIT_ASYNC(n) asm volatile("s_wait_asynccnt %0" ::"i"(n) : "memory")
#endif
#else
#define WAIT_ASYNC(n)
#endif

typedef __attribute__((ext_vector_type(16))) __bf16 v16bf;
typedef __attribute__((ext_vector_type(8)))  __bf16 v8bf;
typedef __attribute__((ext_vector_type(8)))  float  v8f;
typedef __attribute__((ext_vector_type(4)))  float  v4f;

static __device__ __forceinline__ v8f wmma_bf16(v16bf a, v16bf b, v8f c) {
  // D = A(16x32 bf16) * B(32x16 bf16) + C(16x16 f32)
  return __builtin_amdgcn_wmma_f32_16x16x32_bf16(
      /*neg_a=*/false, a, /*neg_b=*/false, b,
      /*c_mod=*/(short)0, c, /*reuse_a=*/false, /*reuse_b=*/false);
}

static __device__ __forceinline__ v16bf pack16(v8bf lo, v8bf hi) {
  return __builtin_shufflevector(lo, hi, 0,1,2,3,4,5,6,7,8,9,10,11,12,13,14,15);
}
static __device__ __forceinline__ v16bf load16(const __bf16* p0, const __bf16* p1) {
  return pack16(*(const v8bf*)p0, *(const v8bf*)p1);
}
static __device__ __forceinline__ v16bf cvt16(v4f a, v4f b, v4f c, v4f d) {
  v16bf r;
#pragma unroll
  for (int i = 0; i < 4; ++i) {
    r[i]      = (__bf16)a[i];
    r[4 + i]  = (__bf16)b[i];
    r[8 + i]  = (__bf16)c[i];
    r[12 + i] = (__bf16)d[i];
  }
  return r;
}

// Reductions across the 16 lanes of a half-wave (C-layout row reduction).
static __device__ __forceinline__ float redmax16(float x) {
  x = fmaxf(x, __shfl_xor(x, 1, 32));
  x = fmaxf(x, __shfl_xor(x, 2, 32));
  x = fmaxf(x, __shfl_xor(x, 4, 32));
  x = fmaxf(x, __shfl_xor(x, 8, 32));
  return x;
}
static __device__ __forceinline__ float redsum16(float x) {
  x += __shfl_xor(x, 1, 32);
  x += __shfl_xor(x, 2, 32);
  x += __shfl_xor(x, 4, 32);
  x += __shfl_xor(x, 8, 32);
  return x;
}

#if HAVE_ASYNC
// Builtin signature (from hipcc diagnostic): pointee is int __vector_size__(16),
// global src in AS1 (printed "__device__"), LDS dst in AS3.
typedef int v4i_t __attribute__((__vector_size__(16)));
typedef v4i_t __attribute__((address_space(1))) * gbl_v4i_p;
typedef v4i_t __attribute__((address_space(3))) * lds_v4i_p;

static __device__ __forceinline__ void async_cp16(const __bf16* g, __bf16* l) {
  __builtin_amdgcn_global_load_async_to_lds_b128(
      (gbl_v4i_p)(g), (lds_v4i_p)(l), 0, 0);
}
#endif

// Stage one 64-key block: K (64x32 bf16) + V (256x64 bf16) -> LDS.
// 9 x 16B transfers per thread (1 K + 8 V), async when available.
static __device__ __forceinline__ void stage_block(
    const __bf16* __restrict__ Kg, const __bf16* __restrict__ Vg,
    __bf16* dK, __bf16* dV, int tid) {
  const int krow = tid >> 2, kpart = tid & 3;
#if HAVE_ASYNC
  async_cp16(Kg + krow * DD + kpart * 8, dK + krow * KSTR + kpart * 8);
#pragma unroll
  for (int it = 0; it < 8; ++it) {
    int lin = tid + it * 256;
    int ch = lin >> 3, part = lin & 7;
    async_cp16(Vg + (size_t)ch * HW + part * 8, dV + ch * VSTR + part * 8);
  }
#else
  *(uint4*)(dK + krow * KSTR + kpart * 8) =
      *(const uint4*)(Kg + krow * DD + kpart * 8);
#pragma unroll
  for (int it = 0; it < 8; ++it) {
    int lin = tid + it * 256;
    int ch = lin >> 3, part = lin & 7;
    *(uint4*)(dV + ch * VSTR + part * 8) =
        *(const uint4*)(Vg + (size_t)ch * HW + part * 8);
  }
#endif
}

// ---------------------------------------------------------------------------
// Projection: Y[o,p] = (sum_c W[o,c] * X[b,c,p] + bias[o]) * scale, as bf16.
// ---------------------------------------------------------------------------
__global__ __launch_bounds__(128)
void qkv_proj_kernel(const float* __restrict__ X, const float* __restrict__ W,
                     const float* __restrict__ bias, __bf16* __restrict__ Y,
                     int M, int chMajorOut, float scale) {
  __shared__ __bf16 sX[64 * 264];  // 64 positions x 256 channels (+8 pad)

  const int b    = blockIdx.y;
  const int p0   = blockIdx.x * 64;
  const int tid  = threadIdx.x;
  const int wv   = tid >> 5;
  const int lane = tid & 31;
  const int n    = lane & 15;
  const int h    = lane >> 4;

#pragma unroll
  for (int it = 0; it < 32; ++it) {
    int vecid = tid + it * 128;
    int ch = vecid >> 4;
    int pg = vecid & 15;
    v4f xv = *(const v4f*)(X + (size_t)(b * CC + ch) * HW + p0 + pg * 4);
#pragma unroll
    for (int k = 0; k < 4; ++k)
      sX[(pg * 4 + k) * 264 + ch] = (__bf16)xv[k];
  }
  __syncthreads();

  const int nmt = M >> 4;
  for (int mt = 0; mt < nmt; ++mt) {
    v8f acc = {};
#pragma unroll
    for (int k0 = 0; k0 < 256; k0 += 32) {
      const float* wr = W + (size_t)(mt * 16 + n) * 256 + k0 + h * 8;
      v4f a0 = *(const v4f*)(wr);
      v4f a1 = *(const v4f*)(wr + 4);
      v4f a2 = *(const v4f*)(wr + 16);
      v4f a3 = *(const v4f*)(wr + 20);
      v16bf aW = cvt16(a0, a1, a2, a3);
      const __bf16* xp = sX + (wv * 16 + n) * 264 + k0 + h * 8;
      v16bf bX = load16(xp, xp + 16);
      acc = wmma_bf16(aW, bX, acc);
    }
    const int p = p0 + wv * 16 + n;
#pragma unroll
    for (int v = 0; v < 8; ++v) {
      int o = mt * 16 + v + 8 * h;  // C-layout: row = v + 8*half
      float y = (acc[v] + bias[o]) * scale;
      if (chMajorOut)
        Y[(size_t)(b * M + o) * HW + p] = (__bf16)y;
      else
        Y[((size_t)b * HW + p) * M + o] = (__bf16)y;
    }
  }
}

// ---------------------------------------------------------------------------
// Flash attention: 64 queries x 256 channels per workgroup (8 waves =
// 4 query row-tiles x 2 channel halves).  Key blocks of 64 double-buffered.
// Qb/Kb position-major [B][HW][32] (Q pre-scaled by 1/sqrt(32)),
// Vb channel-major [B][256][HW].
// ---------------------------------------------------------------------------
__global__ __launch_bounds__(256)
void attn_kernel(const __bf16* __restrict__ Qb, const __bf16* __restrict__ Kb,
                 const __bf16* __restrict__ Vb, float* __restrict__ out) {
  __shared__ __bf16 sK[2][64 * KSTR];   // [key][d], padded
  __shared__ __bf16 sV[2][CC * VSTR];   // [ch][key], padded
  __shared__ __bf16 sP[8 * 16 * PSTR];  // per-wave P slab, padded

  const int b      = blockIdx.y;
  const int i0     = blockIdx.x * 64;
  const int tid    = threadIdx.x;
  const int wv     = tid >> 5;
  const int lane   = tid & 31;
  const int n      = lane & 15;
  const int h      = lane >> 4;
  const int rowT   = wv >> 1;           // query row tile 0..3
  const int chHalf = wv & 1;            // channel half 0..1

  // Q A-tile (16 queries x 32 d), lane l holds row l%16.
  const __bf16* qrow = Qb + ((size_t)(b * HW + i0 + rowT * 16 + n)) * DD;
  const v16bf aQ = load16(qrow + h * 16, qrow + h * 16 + 8);

  v8f O[8];
  float m[8], lsum[8];
#pragma unroll
  for (int t = 0; t < 8; ++t) O[t] = (v8f){};
#pragma unroll
  for (int v = 0; v < 8; ++v) { m[v] = -1e30f; lsum[v] = 0.0f; }

  __bf16* myP = sP + wv * (16 * PSTR);
  const __bf16* Kgb = Kb + (size_t)b * HW * DD;
  const __bf16* Vgb = Vb + (size_t)b * CC * HW;

  // Prologue: kick off block 0.
  stage_block(Kgb, Vgb, sK[0], sV[0], tid);

  for (int j0 = 0; j0 < HW; j0 += 64) {
    const int cur = (j0 >> 6) & 1;
    __syncthreads();  // all waves done reading buffer cur^1
    if (j0 + 64 < HW) {
      stage_block(Kgb + (size_t)(j0 + 64) * DD, Vgb + (j0 + 64),
                  sK[cur ^ 1], sV[cur ^ 1], tid);
      WAIT_ASYNC(9);  // 9 newer ops pending -> current block's 9 have landed
    } else {
      WAIT_ASYNC(0);
    }
#if !HAVE_ASYNC
    if (j0 + 64 < HW)
      __builtin_prefetch(Vgb + (size_t)tid * HW + j0 + 64, 0, 1);
#endif
    __syncthreads();  // every thread's copies for buffer cur are visible

    const __bf16* sKc = sK[cur];
    const __bf16* sVc = sV[cur];

    // --- S = Q^T K: 16 query rows x 64 keys (4 WMMAs, K = d = 32)
    v8f Sv[4];
#pragma unroll
    for (int t = 0; t < 4; ++t) {
      const __bf16* kp = sKc + (t * 16 + n) * KSTR + h * 16;
      v16bf bK = load16(kp, kp + 8);
      v8f z = {};
      Sv[t] = wmma_bf16(aQ, bK, z);
    }

    // --- online softmax (row stats replicated across the 16 lanes)
    float corr[8];
#pragma unroll
    for (int v = 0; v < 8; ++v) {
      float mx = fmaxf(fmaxf(Sv[0][v], Sv[1][v]), fmaxf(Sv[2][v], Sv[3][v]));
      mx = redmax16(mx);
      float mn = fmaxf(m[v], mx);
      corr[v] = __expf(m[v] - mn);
      m[v] = mn;
    }
#pragma unroll
    for (int t = 0; t < 4; ++t)
#pragma unroll
      for (int v = 0; v < 8; ++v)
        Sv[t][v] = __expf(Sv[t][v] - m[v]);
#pragma unroll
    for (int v = 0; v < 8; ++v) {
      float sm = Sv[0][v] + Sv[1][v] + Sv[2][v] + Sv[3][v];
      sm = redsum16(sm);
      lsum[v] = lsum[v] * corr[v] + sm;
    }
#pragma unroll
    for (int ct = 0; ct < 8; ++ct)
#pragma unroll
      for (int v = 0; v < 8; ++v)
        O[ct][v] *= corr[v];

    // --- P: C-layout -> A-layout via this wave's private LDS slab
#pragma unroll
    for (int t = 0; t < 4; ++t)
#pragma unroll
      for (int v = 0; v < 8; ++v)
        myP[(v + 8 * h) * PSTR + t * 16 + n] = (__bf16)Sv[t][v];
    asm volatile("s_wait_dscnt 0x0" ::: "memory");  // same-wave LDS RAW

    v16bf pA[2];
#pragma unroll
    for (int s = 0; s < 2; ++s) {
      const __bf16* pp = myP + n * PSTR + s * 32 + h * 16;
      pA[s] = load16(pp, pp + 8);
    }

    // --- O += P * V^T  (16 WMMAs: 8 channel tiles x 2 key subtiles)
#pragma unroll
    for (int ct = 0; ct < 8; ++ct) {
      int ch = chHalf * 128 + ct * 16 + n;
      const __bf16* vp0 = sVc + ch * VSTR + h * 16;
      O[ct] = wmma_bf16(pA[0], load16(vp0, vp0 + 8), O[ct]);
      const __bf16* vp1 = sVc + ch * VSTR + 32 + h * 16;
      O[ct] = wmma_bf16(pA[1], load16(vp1, vp1 + 8), O[ct]);
    }
  }

  // --- epilogue: normalize and write out[b][ch][i] (f32)
#pragma unroll
  for (int v = 0; v < 8; ++v) lsum[v] = 1.0f / lsum[v];
#pragma unroll
  for (int ct = 0; ct < 8; ++ct) {
    int ch = chHalf * 128 + ct * 16 + n;
    float* op = out + (size_t)(b * CC + ch) * HW + i0 + rowT * 16 + 8 * h;
#pragma unroll
    for (int v = 0; v < 8; ++v)
      op[v] = O[ct][v] * lsum[v];
  }
}

// ---------------------------------------------------------------------------
extern "C" void kernel_launch(void* const* d_in, const int* in_sizes, int n_in,
                              void* d_out, int out_size, void* d_ws, size_t ws_size,
                              hipStream_t stream) {
  (void)in_sizes; (void)n_in; (void)out_size; (void)ws_size;
  const float* feat1 = (const float*)d_in[0];
  const float* feat2 = (const float*)d_in[1];
  const float* wq    = (const float*)d_in[2];
  const float* bq    = (const float*)d_in[3];
  const float* wk    = (const float*)d_in[4];
  const float* bk    = (const float*)d_in[5];
  const float* wv    = (const float*)d_in[6];
  const float* bv    = (const float*)d_in[7];
  float* out = (float*)d_out;

  // Workspace: Qb [B][HW][32] bf16 (1MB) | Kb same (1MB) | Vb [B][256][HW] (8MB)
  __bf16* Qb = (__bf16*)d_ws;
  __bf16* Kb = Qb + (size_t)BB * HW * DD;
  __bf16* Vb = Kb + (size_t)BB * HW * DD;

  const float scale = 0.17677669529663687f;  // 1/sqrt(32), folded into Q

  dim3 gp(HW / 64, BB);
  qkv_proj_kernel<<<gp, 128, 0, stream>>>(feat1, wq, bq, Qb, DD, 0, scale);
  qkv_proj_kernel<<<gp, 128, 0, stream>>>(feat2, wk, bk, Kb, DD, 0, 1.0f);
  qkv_proj_kernel<<<gp, 128, 0, stream>>>(feat2, wv, bv, Vb, CC, 1, 1.0f);

  attn_kernel<<<dim3(HW / 64, BB), 256, 0, stream>>>(Qb, Kb, Vb, out);
}
